// QuantizedLinear_11562051961558
// MI455X (gfx1250) — compile-verified, matrix operand order checked by reading, not run on hardware
//
#include <hip/hip_runtime.h>
#include <hip/hip_bf16.h>

// ---------------------------------------------------------------------------
// W4 (group-32 affine) x FP32 activations -> bf16 WMMA GEMM for gfx1250.
// M = B*S = 4096, N = 4096, K = 4096, Kp = 512 packed int32 words, G = 128.
// Preferred path: one-shot x fp32->bf16 prepass into d_ws (32 MB), then a
// bf16 x bf16 WMMA GEMM whose x-staging is pure copies (no per-iter cvt).
// Fallback path (ws too small): fused conversion (round-1/2 kernel).
// Block tile 128x128, BK = 64, 256 threads (8 waves), wave tile 64x32.
// Double-buffered LDS, software-pipelined global loads, one barrier/iter.
// ---------------------------------------------------------------------------

typedef __attribute__((ext_vector_type(16))) __bf16 v16bf;
typedef __attribute__((ext_vector_type(8)))  __bf16 v8bf;
typedef __attribute__((ext_vector_type(8)))  float  v8f;

#define M_DIM 4096
#define N_DIM 4096
#define K_DIM 4096
#define KP    512      // packed int32 words per output row
#define G_DIM 128      // groups per row (group_size = 32)

#define BM 128
#define BN 128
#define BK 64
#define NT (K_DIM / BK)   // 64 K-tiles
#define LDA 72            // padded LDS row stride in bf16 (144 B = 9*16 B, bank-stride 36)

// Pack two f32 -> two bf16 in one v_perm_b32 (plus cheap round-bias adds).
static __device__ __forceinline__ unsigned pack_bf16x2(float a, float b) {
    unsigned ua = __builtin_bit_cast(unsigned, a) + 0x8000u;
    unsigned ub = __builtin_bit_cast(unsigned, b) + 0x8000u;
    // result[15:0] = ua[31:16], result[31:16] = ub[31:16]
    return __builtin_amdgcn_perm(ub, ua, 0x07060302u);
}

union FragBF { v16bf v; v8bf h[2]; };

// ---------------- shared weight staging ------------------------------------
struct WStage { int4 wv; float s, z; };

__device__ __forceinline__ void load_w(const int* __restrict__ qPtr,
                                       const float* __restrict__ sPtr,
                                       const float* __restrict__ zPtr,
                                       int kt, WStage& r) {
    r.wv = *(const int4*)(qPtr + (size_t)kt * 8);
    r.s  = sPtr[kt * 2];
    r.z  = zPtr[kt * 2];
}

__device__ __forceinline__ void store_w(__bf16* __restrict__ Bsb,
                                        const WStage& r, int wqn, int wqg) {
    const float o = -r.z * r.s;                    // w = q*s + o
    const unsigned wd[4] = {(unsigned)r.wv.x, (unsigned)r.wv.y,
                            (unsigned)r.wv.z, (unsigned)r.wv.w};
    #pragma unroll
    for (int j = 0; j < 4; ++j) {
        uint4 ow;
        unsigned* owp = (unsigned*)&ow;
        #pragma unroll
        for (int u = 0; u < 4; ++u) {
            const float q0 = (float)((wd[j] >> (8 * u))     & 15u);
            const float q1 = (float)((wd[j] >> (8 * u + 4)) & 15u);
            owp[u] = pack_bf16x2(fmaf(q0, r.s, o), fmaf(q1, r.s, o));
        }
        *(uint4*)(&Bsb[wqn * LDA + wqg * 32 + j * 8]) = ow;
    }
}

// ---------------- shared compute + epilogue --------------------------------
__device__ __forceinline__ void compute_tile(const __bf16* __restrict__ Asb,
                                             const __bf16* __restrict__ Bsb,
                                             v8f acc[4][2],
                                             int lane, int wm, int wn) {
    #pragma unroll
    for (int kb = 0; kb < 2; ++kb) {
        const int kOff = kb * 32 + (lane >> 4) * 8;

        FragBF afrag[4];
        #pragma unroll
        for (int mt = 0; mt < 4; ++mt) {
            const __bf16* p = &Asb[(wm + mt * 16 + (lane & 15)) * LDA + kOff];
            afrag[mt].h[0] = *(const v8bf*)(p);
            afrag[mt].h[1] = *(const v8bf*)(p + 16);
        }
        FragBF bfrag[2];
        #pragma unroll
        for (int nt = 0; nt < 2; ++nt) {
            const __bf16* p = &Bsb[(wn + nt * 16 + (lane & 15)) * LDA + kOff];
            bfrag[nt].h[0] = *(const v8bf*)(p);
            bfrag[nt].h[1] = *(const v8bf*)(p + 16);
        }
        #pragma unroll
        for (int mt = 0; mt < 4; ++mt)
            #pragma unroll
            for (int nt = 0; nt < 2; ++nt)
                acc[mt][nt] = __builtin_amdgcn_wmma_f32_16x16x32_bf16(
                    false, afrag[mt].v, false, bfrag[nt].v,
                    (short)0, acc[mt][nt], false, false);
    }
}

__device__ __forceinline__ void epilogue(v8f acc[4][2],
                                         const float* __restrict__ bias,
                                         float* __restrict__ out,
                                         int mBlock, int nBlock,
                                         int wm, int wn, int lane) {
    #pragma unroll
    for (int nt = 0; nt < 2; ++nt) {
        const int col = nBlock + wn + nt * 16 + (lane & 15);
        const float bv = bias[col];
        #pragma unroll
        for (int mt = 0; mt < 4; ++mt) {
            const int rbase = mBlock + wm + mt * 16 + (lane >> 4) * 8;
            #pragma unroll
            for (int v = 0; v < 8; ++v)
                out[(size_t)(rbase + v) * N_DIM + col] = acc[mt][nt][v] + bv;
        }
    }
}

// ---------------- prepass: x fp32 -> bf16 (once) ---------------------------
__global__ __launch_bounds__(256)
void convert_x_bf16(const float* __restrict__ x, __bf16* __restrict__ xb) {
    const size_t i = ((size_t)blockIdx.x * 256 + threadIdx.x) * 8;
    const float4 a = *(const float4*)(x + i);
    const float4 b = *(const float4*)(x + i + 4);
    uint4 o;
    o.x = pack_bf16x2(a.x, a.y);
    o.y = pack_bf16x2(a.z, a.w);
    o.z = pack_bf16x2(b.x, b.y);
    o.w = pack_bf16x2(b.z, b.w);
    *(uint4*)(xb + i) = o;
}

// ---------------- preferred GEMM: bf16 x from workspace --------------------
__global__ __launch_bounds__(256)
void w4a16_wmma_gemm(const __bf16* __restrict__ xb,
                     const int*    __restrict__ qw,
                     const float*  __restrict__ scale,
                     const float*  __restrict__ zp,
                     const float*  __restrict__ bias,
                     float*        __restrict__ out) {
    __shared__ __bf16 As[2][BM * LDA];
    __shared__ __bf16 Bs[2][BN * LDA];

    const int t    = threadIdx.x;
    const int lane = t & 31;
    const int w    = t >> 5;

    const int mBlock = blockIdx.y * BM;
    const int nBlock = blockIdx.x * BN;
    const int wm = (w >> 2) * 64;
    const int wn = (w &  3) * 32;

    // x staging: thread -> (row, half-row of 32 bf16 = 64 B = 4 x uint4)
    const int xrow  = t >> 1;         // 0..127
    const int xhalf = t & 1;          // 0..1
    // weight staging: thread -> (n, 32-wide K group)
    const int wqn = t >> 1;           // 0..127
    const int wqg = t & 1;            // 0..1

    const __bf16* xPtr = xb    + (size_t)(mBlock + xrow) * K_DIM + xhalf * 32;
    const int*    qPtr = qw    + (size_t)(nBlock + wqn)  * KP    + wqg * 4;
    const float*  sPtr = scale + (size_t)(nBlock + wqn)  * G_DIM + wqg;
    const float*  zPtr = zp    + (size_t)(nBlock + wqn)  * G_DIM + wqg;

    v8f acc[4][2] = {};

    // prologue: stage tile 0
    {
        uint4 xq[4];
        #pragma unroll
        for (int q = 0; q < 4; ++q) xq[q] = *(const uint4*)(xPtr + q * 8);
        WStage ws; load_w(qPtr, sPtr, zPtr, 0, ws);
        #pragma unroll
        for (int q = 0; q < 4; ++q)
            *(uint4*)(&As[0][xrow * LDA + xhalf * 32 + q * 8]) = xq[q];
        store_w(Bs[0], ws, wqn, wqg);
    }
    __syncthreads();

    for (int kt = 0; kt < NT - 1; ++kt) {
        uint4 xq[4];
        #pragma unroll
        for (int q = 0; q < 4; ++q)
            xq[q] = *(const uint4*)(xPtr + (kt + 1) * BK + q * 8);
        WStage ws; load_w(qPtr, sPtr, zPtr, kt + 1, ws);

        compute_tile(As[kt & 1], Bs[kt & 1], acc, lane, wm, wn);

        __bf16* Asb = As[(kt + 1) & 1];
        #pragma unroll
        for (int q = 0; q < 4; ++q)
            *(uint4*)(&Asb[xrow * LDA + xhalf * 32 + q * 8]) = xq[q];
        store_w(Bs[(kt + 1) & 1], ws, wqn, wqg);
        __syncthreads();
    }
    compute_tile(As[(NT - 1) & 1], Bs[(NT - 1) & 1], acc, lane, wm, wn);

    epilogue(acc, bias, out, mBlock, nBlock, wm, wn, lane);
}

// ---------------- fallback GEMM: fused fp32-x conversion -------------------
__global__ __launch_bounds__(256)
void w4a32_wmma_gemm(const float* __restrict__ x,
                     const int*   __restrict__ qw,
                     const float* __restrict__ scale,
                     const float* __restrict__ zp,
                     const float* __restrict__ bias,
                     float*       __restrict__ out) {
    __shared__ __bf16 As[2][BM * LDA];
    __shared__ __bf16 Bs[2][BN * LDA];

    const int t    = threadIdx.x;
    const int lane = t & 31;
    const int w    = t >> 5;

    const int mBlock = blockIdx.y * BM;
    const int nBlock = blockIdx.x * BN;
    const int wm = (w >> 2) * 64;
    const int wn = (w &  3) * 32;

    const int xr  = t >> 4;           // 0..15 row-in-pass (x)
    const int xc  = t & 15;           // 0..15 float4-in-row (x)
    const int wqn = t >> 1;
    const int wqg = t & 1;

    const float* xPtr = x     + (size_t)(mBlock + xr)  * K_DIM + xc * 4;
    const int*   qPtr = qw    + (size_t)(nBlock + wqn) * KP    + wqg * 4;
    const float* sPtr = scale + (size_t)(nBlock + wqn) * G_DIM + wqg;
    const float* zPtr = zp    + (size_t)(nBlock + wqn) * G_DIM + wqg;

    v8f acc[4][2] = {};

    {
        float4 xf[8];
        #pragma unroll
        for (int p = 0; p < 8; ++p)
            xf[p] = *(const float4*)(xPtr + (size_t)p * 16 * K_DIM);
        WStage ws; load_w(qPtr, sPtr, zPtr, 0, ws);
        #pragma unroll
        for (int p = 0; p < 8; ++p) {
            uint2 pk;
            pk.x = pack_bf16x2(xf[p].x, xf[p].y);
            pk.y = pack_bf16x2(xf[p].z, xf[p].w);
            *(uint2*)(&As[0][(p * 16 + xr) * LDA + xc * 4]) = pk;
        }
        store_w(Bs[0], ws, wqn, wqg);
    }
    __syncthreads();

    for (int kt = 0; kt < NT - 1; ++kt) {
        float4 xf[8];
        #pragma unroll
        for (int p = 0; p < 8; ++p)
            xf[p] = *(const float4*)(xPtr + (size_t)p * 16 * K_DIM + (kt + 1) * BK);
        WStage ws; load_w(qPtr, sPtr, zPtr, kt + 1, ws);

        compute_tile(As[kt & 1], Bs[kt & 1], acc, lane, wm, wn);

        __bf16* Asb = As[(kt + 1) & 1];
        #pragma unroll
        for (int p = 0; p < 8; ++p) {
            uint2 pk;
            pk.x = pack_bf16x2(xf[p].x, xf[p].y);
            pk.y = pack_bf16x2(xf[p].z, xf[p].w);
            *(uint2*)(&Asb[(p * 16 + xr) * LDA + xc * 4]) = pk;
        }
        store_w(Bs[(kt + 1) & 1], ws, wqn, wqg);
        __syncthreads();
    }
    compute_tile(As[(NT - 1) & 1], Bs[(NT - 1) & 1], acc, lane, wm, wn);

    epilogue(acc, bias, out, mBlock, nBlock, wm, wn, lane);
}

extern "C" void kernel_launch(void* const* d_in, const int* in_sizes, int n_in,
                              void* d_out, int out_size, void* d_ws, size_t ws_size,
                              hipStream_t stream) {
    const float* x     = (const float*)d_in[0];   // [2,2048,4096] fp32
    const int*   qw    = (const int*)  d_in[1];   // [4096,512]    int32 packed 4-bit
    const float* scale = (const float*)d_in[2];   // [4096,128]    fp32
    const float* zp    = (const float*)d_in[3];   // [4096,128]    fp32
    const float* bias  = (const float*)d_in[4];   // [4096]        fp32
    float* out = (float*)d_out;                   // [2,2048,4096] fp32

    dim3 grid(N_DIM / BN, M_DIM / BM);            // 32 x 32 blocks
    dim3 block(256);                              // 8 waves (wave32)

    const size_t xbf_bytes = (size_t)M_DIM * K_DIM * 2;   // 32 MB
    if (ws_size >= xbf_bytes) {
        __bf16* xb = (__bf16*)d_ws;
        const int nconv = (M_DIM * K_DIM) / (256 * 8);    // 8192 blocks
        convert_x_bf16<<<nconv, 256, 0, stream>>>(x, xb);
        w4a16_wmma_gemm<<<grid, block, 0, stream>>>(xb, qw, scale, zp, bias, out);
    } else {
        w4a32_wmma_gemm<<<grid, block, 0, stream>>>(x, qw, scale, zp, bias, out);
    }
}
